// BasicTransformerBlock_48258252538235
// MI455X (gfx1250) — compile-verified, matrix operand order checked by reading, not run on hardware
//
#include <hip/hip_runtime.h>
#include <math.h>

typedef _Float16 f16;
typedef _Float16 v8h  __attribute__((ext_vector_type(8)));
typedef _Float16 v16h __attribute__((ext_vector_type(16)));
typedef float    v8f  __attribute__((ext_vector_type(8)));

#define DEV __device__ __forceinline__

DEV v16h cat8(v8h lo, v8h hi) {
  return __builtin_shufflevector(lo, hi, 0,1,2,3,4,5,6,7,8,9,10,11,12,13,14,15);
}

// D = A(16x32 f16) * B(32x16 f16) + C(16x16 f32)
DEV v8f wmma32(v16h a, v16h b, v8f c) {
  return __builtin_amdgcn_wmma_f32_16x16x32_f16(false, a, false, b, (short)0, c,
                                                false, false);
}

// ---------------------------------------------------------------------------
// Fragment loaders.
// A 16x32 f16 fragment layout (wave32):
//   lane L<16 : row m0+L, halves[0..7]=K k0..k0+7, halves[8..15]=K k0+16..k0+23
//   lane L>=16: row m0+L-16, halves[0..7]=K k0+8..k0+15, halves[8..15]=k0+24..31
// B 32x16 layout is symmetric with N<->lane and K<->halves (column-major cols);
// weights are pre-packed so each lane reads 32 contiguous bytes per B tile.
// ---------------------------------------------------------------------------
template<bool GUARD>
DEV v16h load_a(const f16* A, int lda, int M, int m0, int k0) {
  int lane = threadIdx.x & 31;
  int row = m0 + (lane & 15);
  int kb  = k0 + ((lane & 16) ? 8 : 0);
  v8h lo = {}, hi = {};
  if (!GUARD || row < M) {
    lo = *(const v8h*)(A + (size_t)row * lda + kb);
    hi = *(const v8h*)(A + (size_t)row * lda + kb + 16);
  }
  return cat8(lo, hi);
}

DEV v16h load_bp(const f16* Bp, int KT, int kt, int nt) {
  int lane = threadIdx.x & 31;
  const f16* p = Bp + (((size_t)nt * KT + kt) * 32 + lane) * 16;
  return cat8(*(const v8h*)p, *(const v8h*)(p + 8));
}

// Head fragments for attention: d_head = 40, zero-padded to K=64 (2 k-steps).
DEV v16h load_head_a(const f16* base, int ld, int r0, int dstep) {
  int lane = threadIdx.x & 31;
  int row = r0 + (lane & 15);
  int kb = (lane & 16) ? 8 : 0;
  int dlo = dstep * 32 + kb;
  int dhi = dlo + 16;
  v8h lo = {}, hi = {};
  if (dlo < 40) lo = *(const v8h*)(base + (size_t)row * ld + dlo);
  if (dhi < 40) hi = *(const v8h*)(base + (size_t)row * ld + dhi);
  return cat8(lo, hi);
}

template<bool FULL>
DEV v16h load_head_bk(const f16* base, int ld, int j0, int seqK, int dstep) {
  int lane = threadIdx.x & 31;
  int key = j0 + (lane & 15);
  int kb = (lane & 16) ? 8 : 0;
  v8h lo = {}, hi = {};
  if (FULL || key < seqK) {
    int dlo = dstep * 32 + kb;
    int dhi = dlo + 16;
    if (dlo < 40) lo = *(const v8h*)(base + (size_t)key * ld + dlo);
    if (dhi < 40) hi = *(const v8h*)(base + (size_t)key * ld + dhi);
  }
  return cat8(lo, hi);
}

// ---------------------------------------------------------------------------
// Generic WMMA GEMM: C[M,N] = A[M,K] * B[K,N] (+bias) (+res) -> f16 or f32.
// Block 256 threads (8 waves), macro-tile 128x64; each wave does 32x32.
// GUARD=false when M % 128 == 0 -> straight-line b128 loads in the hot loop.
// ---------------------------------------------------------------------------
template<bool BIAS, bool RES, bool OUTF16, bool GUARD>
__global__ __launch_bounds__(256) void gemm_kernel(
    const f16* __restrict__ A, const f16* __restrict__ Bp,
    const float* __restrict__ bias, const float* res,
    void* outp, int M, int N, int K)
{
  const int KT = K >> 5;
  const int wave = threadIdx.x >> 5;
  const int wm = wave & 3, wn = wave >> 2;
  const int m0 = blockIdx.x * 128 + wm * 32;
  const int n0 = blockIdx.y * 64 + wn * 32;
  const int lane = threadIdx.x & 31;
  const size_t arow0 = (size_t)(m0 + (lane & 15)) * K + ((lane & 16) ? 8 : 0);

  v8f acc[2][2] = {};
  for (int kt = 0; kt < KT; ++kt) {
    if (!GUARD && kt + 1 < KT)        // stream next A k-tile into cache
      __builtin_prefetch(A + arow0 + (kt + 1) * 32, 0, 0);
    v16h a0 = load_a<GUARD>(A, K, M, m0, kt * 32);
    v16h a1 = load_a<GUARD>(A, K, M, m0 + 16, kt * 32);
    v16h b0 = load_bp(Bp, KT, kt, (n0 >> 4));
    v16h b1 = load_bp(Bp, KT, kt, (n0 >> 4) + 1);
    acc[0][0] = wmma32(a0, b0, acc[0][0]);
    acc[0][1] = wmma32(a0, b1, acc[0][1]);
    acc[1][0] = wmma32(a1, b0, acc[1][0]);
    acc[1][1] = wmma32(a1, b1, acc[1][1]);
  }

  const int nIdx = lane & 15;
  const int mOff = (lane & 16) ? 8 : 0;
  #pragma unroll
  for (int mi = 0; mi < 2; ++mi) {
    #pragma unroll
    for (int ni = 0; ni < 2; ++ni) {
      int col = n0 + ni * 16 + nIdx;
      #pragma unroll
      for (int r = 0; r < 8; ++r) {
        int row = m0 + mi * 16 + mOff + r;
        if (!GUARD || row < M) {
          float v = acc[mi][ni][r];
          if (BIAS) v += bias[col];
          if (RES)  v += res[(size_t)row * N + col];
          if (OUTF16) ((f16*)outp)[(size_t)row * N + col] = (f16)v;
          else        ((float*)outp)[(size_t)row * N + col] = v;
        }
      }
    }
  }
}

// ---------------------------------------------------------------------------
// Flash attention: one wave per (16-query tile, head, batch).
// Q,K,V,O: [B*seq, 320] f16, head h at columns h*40..h*40+39. K padded to 64.
// V is staged into LDS TRANSPOSED ([d][key], row stride 40 halves) so each
// V B-fragment half is one 16B-aligned contiguous ds_load_b128.
// FULL = (seqK % 32 == 0): no key-bounds guards or masking in the hot loop
// (self-attention, 128 iterations); cross-attention (seqK=77) keeps guards.
// ---------------------------------------------------------------------------
template<bool FULL>
__global__ __launch_bounds__(32) void attn_kernel(
    const f16* __restrict__ Q, const f16* __restrict__ Km,
    const f16* __restrict__ Vm, f16* __restrict__ O,
    int seqQ, int seqK, float scale)
{
  __shared__ __align__(16) f16 vt_lds[48][40];  // [d][key] transposed V tile
  __shared__ __align__(16) f16 p_lds[16][32];   // P tile C->A re-layout
  const int lane = threadIdx.x & 31;
  const int q0 = blockIdx.x * 16;
  const int h  = blockIdx.y;
  const int bb = blockIdx.z;
  const f16* Qh = Q  + (size_t)bb * seqQ * 320 + h * 40;
  const f16* Kh = Km + (size_t)bb * seqK * 320 + h * 40;
  const f16* Vh = Vm + (size_t)bb * seqK * 320 + h * 40;
  f16*       Oh = O  + (size_t)bb * seqQ * 320 + h * 40;

  v16h aq0 = load_head_a(Qh, 320, q0, 0);
  v16h aq1 = load_head_a(Qh, 320, q0, 1);

  const int nIdx = lane & 15;
  const int mOff = (lane & 16) ? 8 : 0;
  const int kb   = (lane & 16) ? 8 : 0;

  float mstat[8], lstat[8];
  v8f o0 = {}, o1 = {}, o2 = {};
  #pragma unroll
  for (int r = 0; r < 8; ++r) { mstat[r] = -1e30f; lstat[r] = 0.f; }

  for (int j0 = 0; j0 < seqK; j0 += 32) {
    // S = Q K^T for 32 keys (two 16x16 tiles, 2 k-steps each over padded d=64)
    v8f s0 = {}, s1 = {};
    s0 = wmma32(aq0, load_head_bk<FULL>(Kh, 320, j0, seqK, 0), s0);
    s0 = wmma32(aq1, load_head_bk<FULL>(Kh, 320, j0, seqK, 1), s0);
    s1 = wmma32(aq0, load_head_bk<FULL>(Kh, 320, j0 + 16, seqK, 0), s1);
    s1 = wmma32(aq1, load_head_bk<FULL>(Kh, 320, j0 + 16, seqK, 1), s1);

    const bool ok0 = FULL || (j0 + nIdx) < seqK;   // key uniform over regs
    const bool ok1 = FULL || (j0 + 16 + nIdx) < seqK;
    float p0[8], p1[8];
    #pragma unroll
    for (int r = 0; r < 8; ++r) {
      float x0 = ok0 ? s0[r] * scale : -1e30f;
      float x1 = ok1 ? s1[r] * scale : -1e30f;
      float mx = fmaxf(x0, x1);
      #pragma unroll
      for (int off = 1; off < 16; off <<= 1)
        mx = fmaxf(mx, __shfl_xor(mx, off, 32));   // row spans 16 lanes
      float mn = fmaxf(mstat[r], mx);
      float alpha = __expf(mstat[r] - mn);
      float e0 = __expf(x0 - mn);
      float e1 = __expf(x1 - mn);
      float ps = e0 + e1;
      #pragma unroll
      for (int off = 1; off < 16; off <<= 1)
        ps += __shfl_xor(ps, off, 32);
      lstat[r] = lstat[r] * alpha + ps;
      mstat[r] = mn;
      o0[r] *= alpha; o1[r] *= alpha; o2[r] *= alpha;
      p0[r] = e0; p1[r] = e1;
    }

    __syncthreads();  // WAR vs previous iteration's LDS reads
    {   // stage V tile transposed: lane owns key j0+lane, scatters 48 b16s
      int key = j0 + lane;
      v8h z = {};
      #pragma unroll
      for (int part = 0; part < 5; ++part) {
        v8h val = (FULL || key < seqK)
                      ? *(const v8h*)(Vh + (size_t)key * 320 + part * 8)
                      : z;
        #pragma unroll
        for (int e = 0; e < 8; ++e)
          vt_lds[part * 8 + e][lane] = val[e];
      }
      #pragma unroll
      for (int e = 0; e < 8; ++e)
        vt_lds[40 + e][lane] = (f16)0.f;          // zero pad d 40..47
    }
    #pragma unroll
    for (int r = 0; r < 8; ++r) {   // P: C-layout -> LDS [m][k]
      p_lds[mOff + r][nIdx]      = (f16)p0[r];
      p_lds[mOff + r][16 + nIdx] = (f16)p1[r];
    }
    __syncthreads();

    // P as A-fragment (16x32); V columns as contiguous b128 B-fragment halves
    v16h ap = cat8(*(const v8h*)&p_lds[lane & 15][kb],
                   *(const v8h*)&p_lds[lane & 15][16 + kb]);
    #pragma unroll
    for (int dt = 0; dt < 3; ++dt) {
      int d = dt * 16 + nIdx;
      v16h bv = cat8(*(const v8h*)&vt_lds[d][kb],
                     *(const v8h*)&vt_lds[d][16 + kb]);
      if (dt == 0)      o0 = wmma32(ap, bv, o0);
      else if (dt == 1) o1 = wmma32(ap, bv, o1);
      else              o2 = wmma32(ap, bv, o2);
    }
  }

  #pragma unroll
  for (int r = 0; r < 8; ++r) {
    float inv = 1.f / lstat[r];
    size_t rowBase = (size_t)(q0 + mOff + r) * 320;
    Oh[rowBase + nIdx]      = (f16)(o0[r] * inv);
    Oh[rowBase + 16 + nIdx] = (f16)(o1[r] * inv);
    if (nIdx < 8)
      Oh[rowBase + 32 + nIdx] = (f16)(o2[r] * inv);
  }
}

// ---------------------------------------------------------------------------
// LayerNorm over 320 cols, one wave per row, f32 in -> f16 out.
// ---------------------------------------------------------------------------
__global__ __launch_bounds__(256) void ln_kernel(
    const float* __restrict__ x, const float* __restrict__ w,
    const float* __restrict__ b, f16* __restrict__ out, int rows)
{
  int wave = threadIdx.x >> 5, lane = threadIdx.x & 31;
  int row = blockIdx.x * 8 + wave;
  if (row >= rows) return;
  const float* xr = x + (size_t)row * 320;
  float v[10], s = 0.f, ss = 0.f;
  #pragma unroll
  for (int i = 0; i < 10; ++i) {
    v[i] = xr[lane + 32 * i];
    s += v[i]; ss += v[i] * v[i];
  }
  #pragma unroll
  for (int off = 1; off < 32; off <<= 1) {
    s  += __shfl_xor(s, off, 32);
    ss += __shfl_xor(ss, off, 32);
  }
  float mu  = s * (1.f / 320.f);
  float var = ss * (1.f / 320.f) - mu * mu;
  float rs  = rsqrtf(var + 1e-5f);
  f16* orow = out + (size_t)row * 320;
  #pragma unroll
  for (int i = 0; i < 10; ++i) {
    int c = lane + 32 * i;
    orow[c] = (f16)((v[i] - mu) * rs * w[c] + b[c]);
  }
}

// ---------------------------------------------------------------------------
// Pack f32 weight W[K, ldW] (cols colOff..colOff+N-1) into WMMA B layout:
// element index = (((nt*KT + kt)*32 + lane)*16 + h); per-lane 32B contiguous.
// ---------------------------------------------------------------------------
__global__ __launch_bounds__(256) void pack_b_kernel(
    const float* __restrict__ W, f16* __restrict__ Bp,
    int K, int N, int ldW, int colOff)
{
  int idx = blockIdx.x * 256 + threadIdx.x;
  if (idx >= K * N) return;
  int h    = idx & 15;
  int lane = (idx >> 4) & 31;
  int t    = idx >> 9;
  int KT = K >> 5;
  int kt = t % KT, nt = t / KT;
  int n = nt * 16 + (lane & 15);
  int kk = (lane < 16) ? ((h < 8) ? h : (8 + h))
                       : ((h < 8) ? (8 + h) : (16 + h));
  int k = kt * 32 + kk;
  Bp[idx] = (f16)W[(size_t)k * ldW + colOff + n];
}

__global__ __launch_bounds__(256) void cvt_kernel(
    const float* __restrict__ in, f16* __restrict__ out, size_t n)
{
  size_t i = (size_t)blockIdx.x * 256 + threadIdx.x;
  if (i < n) out[i] = (f16)in[i];
}

// g = u * gelu_exact(gate), in-place capable (out may equal u)
__global__ __launch_bounds__(256) void geglu_kernel(
    const f16* u, const f16* __restrict__ g, f16* out, size_t n)
{
  size_t i = (size_t)blockIdx.x * 256 + threadIdx.x;
  if (i < n) {
    float uu = (float)u[i];
    float gg = (float)g[i];
    float ge = 0.5f * gg * (1.f + erff(gg * 0.70710678118654752f));
    out[i] = (f16)(uu * ge);
  }
}

// ---------------------------------------------------------------------------
// Host orchestration
// ---------------------------------------------------------------------------
static inline size_t alignup(size_t x) { return (x + 255) & ~(size_t)255; }
static inline int cdiv(int a, int b) { return (a + b - 1) / b; }

extern "C" void kernel_launch(void* const* d_in, const int* in_sizes, int n_in,
                              void* d_out, int out_size, void* d_ws, size_t ws_size,
                              hipStream_t stream)
{
  (void)in_sizes; (void)n_in; (void)out_size; (void)ws_size;
  const float* x    = (const float*)d_in[0];
  const float* ctx  = (const float*)d_in[1];
  const float* ln1w = (const float*)d_in[2];
  const float* ln1b = (const float*)d_in[3];
  const float* ln2w = (const float*)d_in[4];
  const float* ln2b = (const float*)d_in[5];
  const float* ln3w = (const float*)d_in[6];
  const float* ln3b = (const float*)d_in[7];
  const float* a1Wq = (const float*)d_in[8];
  const float* a1Wk = (const float*)d_in[9];
  const float* a1Wv = (const float*)d_in[10];
  const float* a1Wo = (const float*)d_in[11];
  const float* a1bo = (const float*)d_in[12];
  const float* a2Wq = (const float*)d_in[13];
  const float* a2Wk = (const float*)d_in[14];
  const float* a2Wv = (const float*)d_in[15];
  const float* a2Wo = (const float*)d_in[16];
  const float* a2bo = (const float*)d_in[17];
  const float* ffW1 = (const float*)d_in[18];
  const float* ffb1 = (const float*)d_in[19];
  const float* ffW2 = (const float*)d_in[20];
  const float* ffb2 = (const float*)d_in[21];

  const int B = 2, SEQ = 4096, ROWS = B * SEQ;   // 8192
  const int DIM = 320, CTXD = 768, CTXROWS = B * 77;  // 154
  const int FF = 1280;
  const float SCALE = 0.15811388300841897f;      // 40^-0.5

  char* base = (char*)d_ws; size_t off = 0;
  auto carve = [&](size_t bytes) -> void* {
    off = alignup(off);
    void* r = base + off;
    off += bytes;
    return r;
  };
  f16* wq1  = (f16*)carve((size_t)DIM * DIM * 2);
  f16* wk1  = (f16*)carve((size_t)DIM * DIM * 2);
  f16* wv1  = (f16*)carve((size_t)DIM * DIM * 2);
  f16* wo1  = (f16*)carve((size_t)DIM * DIM * 2);
  f16* wq2  = (f16*)carve((size_t)DIM * DIM * 2);
  f16* wk2  = (f16*)carve((size_t)CTXD * DIM * 2);
  f16* wv2  = (f16*)carve((size_t)CTXD * DIM * 2);
  f16* wo2  = (f16*)carve((size_t)DIM * DIM * 2);
  f16* wf1u = (f16*)carve((size_t)DIM * FF * 2);
  f16* wf1g = (f16*)carve((size_t)DIM * FF * 2);
  f16* wf2  = (f16*)carve((size_t)FF * DIM * 2);
  f16* ctxh = (f16*)carve((size_t)CTXROWS * CTXD * 2);
  f16* hbuf = (f16*)carve((size_t)ROWS * DIM * 2);
  f16* qbuf = (f16*)carve((size_t)ROWS * DIM * 2);
  f16* kbuf = (f16*)carve((size_t)ROWS * DIM * 2);
  f16* vbuf = (f16*)carve((size_t)ROWS * DIM * 2);
  f16* abuf = (f16*)carve((size_t)ROWS * DIM * 2);
  float* resid = (float*)carve((size_t)ROWS * DIM * 4);
  f16* ubuf = (f16*)carve((size_t)ROWS * FF * 2);
  f16* gbuf = (f16*)carve((size_t)ROWS * FF * 2);

  // ---- pack weights to WMMA B layout (f16) ----
  auto pack = [&](const float* W, f16* Bp, int K, int N, int ldW, int co) {
    pack_b_kernel<<<cdiv(K * N, 256), 256, 0, stream>>>(W, Bp, K, N, ldW, co);
  };
  pack(a1Wq, wq1, DIM, DIM, DIM, 0);
  pack(a1Wk, wk1, DIM, DIM, DIM, 0);
  pack(a1Wv, wv1, DIM, DIM, DIM, 0);
  pack(a1Wo, wo1, DIM, DIM, DIM, 0);
  pack(a2Wq, wq2, DIM, DIM, DIM, 0);
  pack(a2Wk, wk2, CTXD, DIM, DIM, 0);
  pack(a2Wv, wv2, CTXD, DIM, DIM, 0);
  pack(a2Wo, wo2, DIM, DIM, DIM, 0);
  pack(ffW1, wf1u, DIM, FF, 2 * FF, 0);
  pack(ffW1, wf1g, DIM, FF, 2 * FF, FF);
  pack(ffW2, wf2, FF, DIM, DIM, 0);
  {
    size_t n = (size_t)CTXROWS * CTXD;
    cvt_kernel<<<cdiv((int)n, 256), 256, 0, stream>>>(ctx, ctxh, n);
  }

  // M % 128 == 0 -> unguarded fast path
  auto gemm_h = [&](const f16* A, const f16* Bp, f16* out, int M, int N, int K,
                    const float* bias) {
    dim3 g(cdiv(M, 128), N / 64);
    bool guard = (M & 127) != 0;
    if (bias) {
      gemm_kernel<true, false, true, false><<<g, 256, 0, stream>>>(
          A, Bp, bias, nullptr, out, M, N, K);
    } else if (guard) {
      gemm_kernel<false, false, true, true><<<g, 256, 0, stream>>>(
          A, Bp, nullptr, nullptr, out, M, N, K);
    } else {
      gemm_kernel<false, false, true, false><<<g, 256, 0, stream>>>(
          A, Bp, nullptr, nullptr, out, M, N, K);
    }
  };
  auto gemm_fr = [&](const f16* A, const f16* Bp, const float* bias,
                     const float* res, float* out, int M, int N, int K) {
    dim3 g(cdiv(M, 128), N / 64);
    gemm_kernel<true, true, false, false><<<g, 256, 0, stream>>>(
        A, Bp, bias, res, out, M, N, K);
  };

  dim3 attnGrid(SEQ / 16, 8, B);

  // ---- 1) self-attention ----
  ln_kernel<<<ROWS / 8, 256, 0, stream>>>(x, ln1w, ln1b, hbuf, ROWS);
  gemm_h(hbuf, wq1, qbuf, ROWS, DIM, DIM, nullptr);
  gemm_h(hbuf, wk1, kbuf, ROWS, DIM, DIM, nullptr);
  gemm_h(hbuf, wv1, vbuf, ROWS, DIM, DIM, nullptr);
  attn_kernel<true><<<attnGrid, 32, 0, stream>>>(qbuf, kbuf, vbuf, abuf,
                                                 SEQ, SEQ, SCALE);
  gemm_fr(abuf, wo1, a1bo, x, resid, ROWS, DIM, DIM);          // resid = attn1 + x

  // ---- 2) cross-attention ----
  ln_kernel<<<ROWS / 8, 256, 0, stream>>>(resid, ln2w, ln2b, hbuf, ROWS);
  gemm_h(hbuf, wq2, qbuf, ROWS, DIM, DIM, nullptr);
  gemm_h(ctxh, wk2, kbuf, CTXROWS, DIM, CTXD, nullptr);        // guarded (M=154)
  gemm_h(ctxh, wv2, vbuf, CTXROWS, DIM, CTXD, nullptr);        // guarded (M=154)
  attn_kernel<false><<<attnGrid, 32, 0, stream>>>(qbuf, kbuf, vbuf, abuf,
                                                  SEQ, 77, SCALE);
  gemm_fr(abuf, wo2, a2bo, resid, resid, ROWS, DIM, DIM);      // resid += attn2

  // ---- 3) GEGLU feed-forward ----
  ln_kernel<<<ROWS / 8, 256, 0, stream>>>(resid, ln3w, ln3b, hbuf, ROWS);
  gemm_h(hbuf, wf1u, ubuf, ROWS, FF, DIM, ffb1);               // u half
  gemm_h(hbuf, wf1g, gbuf, ROWS, FF, DIM, ffb1 + FF);          // gate half
  {
    size_t n = (size_t)ROWS * FF;
    geglu_kernel<<<cdiv((int)n, 256), 256, 0, stream>>>(ubuf, gbuf, ubuf, n);
  }
  gemm_fr(ubuf, wf2, ffb2, resid, (float*)d_out, ROWS, DIM, FF);  // out = ff + resid
}
